// SAGE_20469814133287
// MI455X (gfx1250) — matrix operand order repaired and checked by reading, not compiled, not run on hardware
//
#include <hip/hip_runtime.h>
#include <stdint.h>

#define N_NODES 65536
#define C_FEAT  64   // in == hid == 64

typedef __attribute__((ext_vector_type(16))) _Float16 v16h;
typedef __attribute__((ext_vector_type(8)))  float    v8f;

// ---------------------------------------------------------------------------
// Degree: one thread per edge, hardware fp32 atomic (global_atomic_add_f32)
// ---------------------------------------------------------------------------
__global__ __launch_bounds__(256) void degree_kernel(
    const long long* __restrict__ dst, float* __restrict__ deg, int E) {
  int e = blockIdx.x * blockDim.x + threadIdx.x;
  if (e < E) unsafeAtomicAdd(&deg[(int)dst[e]], 1.0f);
}

// ---------------------------------------------------------------------------
// Scatter-add of source features to destination rows.
// 16 threads per edge, 4 channels each (float4 gather, 4 fp32 atomics).
// Whole working set is L2-resident on MI455X (192 MB), so atomics hit L2.
// ---------------------------------------------------------------------------
__global__ __launch_bounds__(256) void scatter_kernel(
    const float* __restrict__ x, const long long* __restrict__ src,
    const long long* __restrict__ dst, float* __restrict__ agg, int E) {
  int t  = blockIdx.x * blockDim.x + threadIdx.x;
  int e  = t >> 4;
  int cg = (t & 15) * 4;
  if (e >= E) return;
  int s = (int)src[e];
  int d = (int)dst[e];
  float4 v = *(const float4*)(x + (size_t)s * C_FEAT + cg);
  float* p = agg + (size_t)d * C_FEAT + cg;
  unsafeAtomicAdd(p + 0, v.x);
  unsafeAtomicAdd(p + 1, v.y);
  unsafeAtomicAdd(p + 2, v.z);
  unsafeAtomicAdd(p + 3, v.w);
}

// ---------------------------------------------------------------------------
// WMMA fragment loaders (v_wmma_f32_16x16x32_f16 operands, wave32; contiguous k, same order in A and B)
// A 16x32 f16: lane<16 -> row=lane, K = kc*32 + 0..15
//              lane>=16 ->          K = kc*32 + 16..31
// ---------------------------------------------------------------------------
__device__ __forceinline__ v16h load_a_frag(const float* __restrict__ row,
                                            int kc, int half, float scale) {
  const float* p0 = row + kc * 32 + half * 16;
  float4 q0 = *(const float4*)(p0 + 0);
  float4 q1 = *(const float4*)(p0 + 4);
  float4 q2 = *(const float4*)(p0 + 8);
  float4 q3 = *(const float4*)(p0 + 12);
  v16h a;
  a[0]  = (_Float16)(q0.x * scale); a[1]  = (_Float16)(q0.y * scale);
  a[2]  = (_Float16)(q0.z * scale); a[3]  = (_Float16)(q0.w * scale);
  a[4]  = (_Float16)(q1.x * scale); a[5]  = (_Float16)(q1.y * scale);
  a[6]  = (_Float16)(q1.z * scale); a[7]  = (_Float16)(q1.w * scale);
  a[8]  = (_Float16)(q2.x * scale); a[9]  = (_Float16)(q2.y * scale);
  a[10] = (_Float16)(q2.z * scale); a[11] = (_Float16)(q2.w * scale);
  a[12] = (_Float16)(q3.x * scale); a[13] = (_Float16)(q3.y * scale);
  a[14] = (_Float16)(q3.z * scale); a[15] = (_Float16)(q3.w * scale);
  return a;
}

// B 32x16 f16: lane<16 -> col n=lane, K = kc*32 + 0..15
//              lane>=16 ->            K = kc*32 + 16..31
// B[k][n] = W[n][k]  =>  16 contiguous floats of one weight row per lane.
__device__ __forceinline__ v16h load_b_frag(const float* __restrict__ wrow,
                                            int kc, int half) {
  const float* p = wrow + kc * 32 + half * 16;
  float4 q0 = *(const float4*)(p + 0);
  float4 q1 = *(const float4*)(p + 4);
  float4 q2 = *(const float4*)(p + 8);
  float4 q3 = *(const float4*)(p + 12);
  v16h b;
  b[0]  = (_Float16)q0.x; b[1]  = (_Float16)q0.y;
  b[2]  = (_Float16)q0.z; b[3]  = (_Float16)q0.w;
  b[4]  = (_Float16)q1.x; b[5]  = (_Float16)q1.y;
  b[6]  = (_Float16)q1.z; b[7]  = (_Float16)q1.w;
  b[8]  = (_Float16)q2.x; b[9]  = (_Float16)q2.y;
  b[10] = (_Float16)q2.z; b[11] = (_Float16)q2.w;
  b[12] = (_Float16)q3.x; b[13] = (_Float16)q3.y;
  b[14] = (_Float16)q3.z; b[15] = (_Float16)q3.w;
  return b;
}

// ---------------------------------------------------------------------------
// SAGE layer:  out = act( (agg/deg) @ Wl^T + b + xin @ Wr^T )
// One wave32 per 16-row node tile; K=64 -> 2 WMMAs per weight matrix per
// 16-col output tile. Accumulate f32 via v_wmma_f32_16x16x32_f16.
// ---------------------------------------------------------------------------
template<int NOUT, bool RELU>
__global__ __launch_bounds__(256) void sage_layer(
    const float* __restrict__ agg, const float* __restrict__ deg,
    const float* __restrict__ xin,
    const float* __restrict__ Wl, const float* __restrict__ bias,
    const float* __restrict__ Wr, float* __restrict__ out) {
  const int lane = threadIdx.x & 31;
  const int wave = blockIdx.x * (blockDim.x >> 5) + (threadIdx.x >> 5);
  const int m0   = wave * 16;
  const int r    = lane & 15;
  const int half = lane >> 4;
  const int row  = m0 + r;

  const float invd = 1.0f / fmaxf(deg[row], 1.0f);
  const float* arow = agg + (size_t)row * C_FEAT;
  const float* xrow = xin + (size_t)row * C_FEAT;

  v16h aM0 = load_a_frag(arow, 0, half, invd);
  v16h aM1 = load_a_frag(arow, 1, half, invd);
  v16h aX0 = load_a_frag(xrow, 0, half, 1.0f);
  v16h aX1 = load_a_frag(xrow, 1, half, 1.0f);

#pragma unroll
  for (int nt = 0; nt < NOUT / 16; ++nt) {
    const int nbase = nt * 16;
    const float* wl = Wl + (size_t)(nbase + r) * C_FEAT;
    const float* wr = Wr + (size_t)(nbase + r) * C_FEAT;
    v16h bL0 = load_b_frag(wl, 0, half);
    v16h bL1 = load_b_frag(wl, 1, half);
    v16h bR0 = load_b_frag(wr, 0, half);
    v16h bR1 = load_b_frag(wr, 1, half);

    v8f acc = {};
    acc = __builtin_amdgcn_wmma_f32_16x16x32_f16(false, aM0, false, bL0,
                                                 (short)0, acc, false, false);
    acc = __builtin_amdgcn_wmma_f32_16x16x32_f16(false, aM1, false, bL1,
                                                 (short)0, acc, false, false);
    acc = __builtin_amdgcn_wmma_f32_16x16x32_f16(false, aX0, false, bR0,
                                                 (short)0, acc, false, false);
    acc = __builtin_amdgcn_wmma_f32_16x16x32_f16(false, aX1, false, bR1,
                                                 (short)0, acc, false, false);

    const float bv = bias[nbase + r];
#pragma unroll
    for (int v = 0; v < 8; ++v) {
      float val = acc[v] + bv;
      if (RELU) val = fmaxf(val, 0.0f);
      out[(size_t)(m0 + v + 8 * half) * NOUT + nbase + r] = val;
    }
  }
}

// ---------------------------------------------------------------------------
// In-place log_softmax over 32 channels: one wave32 per node, lane = channel.
// ---------------------------------------------------------------------------
__global__ __launch_bounds__(256) void logsoftmax32(float* __restrict__ out) {
  int wave = blockIdx.x * (blockDim.x >> 5) + (threadIdx.x >> 5);
  int lane = threadIdx.x & 31;
  float v = out[(size_t)wave * 32 + lane];
  float m = v;
#pragma unroll
  for (int off = 16; off > 0; off >>= 1) m = fmaxf(m, __shfl_xor(m, off, 32));
  float e = __expf(v - m);
  float s = e;
#pragma unroll
  for (int off = 16; off > 0; off >>= 1) s += __shfl_xor(s, off, 32);
  out[(size_t)wave * 32 + lane] = v - m - __logf(s);
}

// ---------------------------------------------------------------------------
extern "C" void kernel_launch(void* const* d_in, const int* in_sizes, int n_in,
                              void* d_out, int out_size, void* d_ws, size_t ws_size,
                              hipStream_t stream) {
  const float*     x    = (const float*)d_in[0];
  const long long* edge = (const long long*)d_in[1];   // int64 [2, E]
  const float*     W1l  = (const float*)d_in[2];
  const float*     b1   = (const float*)d_in[3];
  const float*     W1r  = (const float*)d_in[4];
  const float*     W2l  = (const float*)d_in[5];
  const float*     b2   = (const float*)d_in[6];
  const float*     W2r  = (const float*)d_in[7];

  const int E = in_sizes[1] / 2;
  const long long* src = edge;
  const long long* dst = edge + E;

  // Workspace: [deg N][agg N*64][h N*64]  (~34 MB)
  float* deg = (float*)d_ws;
  float* agg = deg + N_NODES;
  float* h   = agg + (size_t)N_NODES * C_FEAT;
  float* out = (float*)d_out;

  // Zero deg + agg (graph-capturable async memset)
  hipMemsetAsync(deg, 0, (size_t)(N_NODES + (size_t)N_NODES * C_FEAT) * sizeof(float), stream);

  degree_kernel<<<(E + 255) / 256, 256, 0, stream>>>(dst, deg, E);

  // ---- Layer 1 ----
  scatter_kernel<<<((size_t)E * 16 + 255) / 256, 256, 0, stream>>>(x, src, dst, agg, E);
  sage_layer<64, true><<<N_NODES / 16 / 8, 256, 0, stream>>>(agg, deg, x, W1l, b1, W1r, h);

  // ---- Layer 2 ----
  hipMemsetAsync(agg, 0, (size_t)N_NODES * C_FEAT * sizeof(float), stream);
  scatter_kernel<<<((size_t)E * 16 + 255) / 256, 256, 0, stream>>>(h, src, dst, agg, E);
  sage_layer<32, false><<<N_NODES / 16 / 8, 256, 0, stream>>>(agg, deg, h, W2l, b2, W2r, out);

  logsoftmax32<<<N_NODES / 8, 256, 0, stream>>>(out);
}
